// PAttention_2052994367522
// MI455X (gfx1250) — matrix operand order repaired and checked by reading, not compiled, not run on hardware
//
#include <hip/hip_runtime.h>
#include <hip/hip_bf16.h>
#include <math.h>

// ---------------------------------------------------------------------------
// PAttention, algebraically refactored:
//   s = x @ (Pk @ Wq).T          (32768 x 1024 x 64 GEMM, bf16x3 WMMA)
//   s = gelu(s * 8 / ||s||_row)  (exact GELU, cross-lane reduction)
//   y = s @ Pv                   (32768 x 64 x 1024 GEMM, bf16x3 WMMA)
// ---------------------------------------------------------------------------

typedef __attribute__((ext_vector_type(16))) __bf16 v16bf;
typedef __attribute__((ext_vector_type(8)))  float  v8f;

#define QLEN 32768
#define DDIM 1024
#define NTOK 64

__device__ __forceinline__ v8f wmma_bf16(v16bf a, v16bf b, v8f c) {
    return __builtin_amdgcn_wmma_f32_16x16x32_bf16(
        false, a, false, b, (short)0, c, false, false);
}

// Split 8 fp32 values (two float4) into bf16 hi/lo halves of a v16bf pair.
__device__ __forceinline__ void cvt8(float4 a, float4 b,
                                     v16bf& hi, v16bf& lo, int base) {
    float v[8] = {a.x, a.y, a.z, a.w, b.x, b.y, b.z, b.w};
#pragma unroll
    for (int i = 0; i < 8; ++i) {
        __bf16 h = (__bf16)v[i];
        hi[base + i] = h;
        lo[base + i] = (__bf16)(v[i] - (float)h);
    }
}

// ---------------------------------------------------------------------------
// prep_M: M = Pk @ Wq  (64 x 1024), then split to bf16 hi/lo.
// One block per token row t; 256 threads, 4 output columns each.
// ---------------------------------------------------------------------------
__global__ __launch_bounds__(256) void prep_M(const float* __restrict__ Wq,
                                              const float* __restrict__ Pk,
                                              __bf16* __restrict__ Mhi,
                                              __bf16* __restrict__ Mlo) {
    __shared__ float pk[DDIM];
    const int t = blockIdx.x;
    for (int i = threadIdx.x; i < DDIM; i += 256)
        pk[i] = Pk[(size_t)t * DDIM + i];
    __syncthreads();

    float acc[4] = {0.f, 0.f, 0.f, 0.f};
    const int e0 = threadIdx.x;
    for (int d = 0; d < DDIM; ++d) {
        const float p = pk[d];
        const float* wrow = Wq + (size_t)d * DDIM;
#pragma unroll
        for (int j = 0; j < 4; ++j) acc[j] += p * wrow[e0 + j * 256];
    }
#pragma unroll
    for (int j = 0; j < 4; ++j) {
        const int e = e0 + j * 256;
        const float v = acc[j];
        const __bf16 h = (__bf16)v;
        Mhi[(size_t)t * DDIM + e] = h;
        Mlo[(size_t)t * DDIM + e] = (__bf16)(v - (float)h);
    }
}

// ---------------------------------------------------------------------------
// prep_PvT: PvT[d][t] = Pv[t][d], split to bf16 hi/lo (WMMA-B friendly).
// ---------------------------------------------------------------------------
__global__ __launch_bounds__(256) void prep_PvT(const float* __restrict__ Pv,
                                                __bf16* __restrict__ PvThi,
                                                __bf16* __restrict__ PvTlo) {
    const int idx = blockIdx.x * 256 + threadIdx.x;   // 0 .. 65535
    const int d = idx >> 6;
    const int t = idx & 63;
    const float v = Pv[(size_t)t * DDIM + d];
    const __bf16 h = (__bf16)v;
    PvThi[idx] = h;
    PvTlo[idx] = (__bf16)(v - (float)h);
}

// ---------------------------------------------------------------------------
// Main fused kernel. 256 threads = 8 wave32; each wave owns 16 rows of x.
// ---------------------------------------------------------------------------
__global__ __launch_bounds__(256) void pattn_main(const float* __restrict__ x,
                                                  const __bf16* __restrict__ Mhi,
                                                  const __bf16* __restrict__ Mlo,
                                                  const __bf16* __restrict__ PvThi,
                                                  const __bf16* __restrict__ PvTlo,
                                                  float* __restrict__ y) {
    __shared__ __align__(16) float sTile[8][16][NTOK];   // post-GELU s', 32 KB

    const int wave = threadIdx.x >> 5;
    const int lane = threadIdx.x & 31;
    const int h    = (lane >> 4) & 1;     // half of the wave (K-offset select)
    const int r    = lane & 15;
    const int rowBase = blockIdx.x * 128 + wave * 16;

    // ---------------- Stage 1: s = x @ M.T  (16 x 64, f32 accum) ----------
    v8f c[4] = {v8f{}, v8f{}, v8f{}, v8f{}};
    const float* xrow = x + (size_t)(rowBase + r) * DDIM;

    for (int kk = 0; kk < DDIM / 32; ++kk) {
        const int k0 = kk * 32 + h * 16;
        // A fragment: rows striped per lane, K groups {0..15}/{16..31}
        float4 a0 = *(const float4*)(xrow + k0);
        float4 a1 = *(const float4*)(xrow + k0 + 4);
        float4 b0 = *(const float4*)(xrow + k0 + 8);
        float4 b1 = *(const float4*)(xrow + k0 + 12);
        v16bf ahi, alo;
        cvt8(a0, a1, ahi, alo, 0);
        cvt8(b0, b1, ahi, alo, 8);

#pragma unroll
        for (int nt = 0; nt < 4; ++nt) {
            const size_t boff = (size_t)(nt * 16 + r) * DDIM + kk * 32 + h * 16;
            v16bf bh = *(const v16bf*)(Mhi + boff);
            v16bf bl = *(const v16bf*)(Mlo + boff);
            c[nt] = wmma_bf16(ahi, bh, c[nt]);   // hi*hi
            c[nt] = wmma_bf16(ahi, bl, c[nt]);   // hi*lo
            c[nt] = wmma_bf16(alo, bh, c[nt]);   // lo*hi
        }
    }

    // ---------------- Stage 2: row L2-norm, scale by 8, exact GELU ---------
    // C layout: element rr <-> row rr + 8*h; column (token) = nt*16 + r.
#pragma unroll
    for (int rr = 0; rr < 8; ++rr) {
        float ss = c[0][rr] * c[0][rr] + c[1][rr] * c[1][rr] +
                   c[2][rr] * c[2][rr] + c[3][rr] * c[3][rr];
        ss += __shfl_xor(ss, 1, 32);
        ss += __shfl_xor(ss, 2, 32);
        ss += __shfl_xor(ss, 4, 32);
        ss += __shfl_xor(ss, 8, 32);          // full row sum within 16-lane half
        const float scale = 8.0f / sqrtf(ss); // sqrt(n_tokens)=8
#pragma unroll
        for (int nt = 0; nt < 4; ++nt) {
            const float z = c[nt][rr] * scale;
            const float g = 0.5f * z * (1.0f + erff(z * 0.70710678118654752f));
            sTile[wave][rr + 8 * h][nt * 16 + r] = g;
        }
    }
    __syncthreads();

    // ---------------- Stage 3: y = s' @ Pv  (16 x 1024) --------------------
    // Build A fragments (rows of s', K = token) once; reuse for all 64 d-tiles.
    v16bf sahi[2], salo[2];
#pragma unroll
    for (int kk = 0; kk < 2; ++kk) {
        const float* p = &sTile[wave][r][kk * 32 + h * 16];
        float4 a0 = *(const float4*)(p);
        float4 a1 = *(const float4*)(p + 4);
        float4 b0 = *(const float4*)(p + 8);
        float4 b1 = *(const float4*)(p + 12);
        cvt8(a0, a1, sahi[kk], salo[kk], 0);
        cvt8(b0, b1, sahi[kk], salo[kk], 8);
    }

    for (int d0 = 0; d0 < DDIM; d0 += 16) {
        v8f cy = v8f{};
#pragma unroll
        for (int kk = 0; kk < 2; ++kk) {
            const size_t boff = (size_t)(d0 + r) * NTOK + kk * 32 + h * 16;
            v16bf bh = *(const v16bf*)(PvThi + boff);
            v16bf bl = *(const v16bf*)(PvTlo + boff);
            cy = wmma_bf16(sahi[kk], bh, cy);
            cy = wmma_bf16(sahi[kk], bl, cy);
            cy = wmma_bf16(salo[kk], bh, cy);
        }
#pragma unroll
        for (int rr = 0; rr < 8; ++rr) {
            y[(size_t)(rowBase + rr + 8 * h) * DDIM + d0 + r] = cy[rr];
        }
    }
}

// ---------------------------------------------------------------------------
extern "C" void kernel_launch(void* const* d_in, const int* in_sizes, int n_in,
                              void* d_out, int out_size, void* d_ws, size_t ws_size,
                              hipStream_t stream) {
    const float* x  = (const float*)d_in[0];   // (32768, 1024)
    const float* Wq = (const float*)d_in[1];   // (1024, 1024)
    const float* Pk = (const float*)d_in[2];   // (64, 1024)
    const float* Pv = (const float*)d_in[3];   // (64, 1024)
    float* y = (float*)d_out;                  // (1, 32768, 1024)

    __bf16* Mhi   = (__bf16*)d_ws;             // 64*1024
    __bf16* Mlo   = Mhi   + (size_t)NTOK * DDIM;
    __bf16* PvThi = Mlo   + (size_t)NTOK * DDIM;
    __bf16* PvTlo = PvThi + (size_t)DDIM * NTOK;   // total 512 KB

    prep_M  <<<NTOK, 256, 0, stream>>>(Wq, Pk, Mhi, Mlo);
    prep_PvT<<<(DDIM * NTOK) / 256, 256, 0, stream>>>(Pv, PvThi, PvTlo);
    pattn_main<<<QLEN / 128, 256, 0, stream>>>(x, Mhi, Mlo, PvThi, PvTlo, y);
}
